// W8A16LinearLayer_40303973105954
// MI455X (gfx1250) — matrix operand order repaired and checked by reading, not compiled, not run on hardware
//
#include <hip/hip_runtime.h>

typedef __attribute__((ext_vector_type(16))) _Float16 v16h;
typedef __attribute__((ext_vector_type(8)))  _Float16 v8h;
typedef __attribute__((ext_vector_type(4)))  _Float16 v4h;
typedef __attribute__((ext_vector_type(8)))  float    v8f;

#define BM 128
#define BN 128
#define BK 32
#define SP 40   // padded LDS row stride in halves (80 bytes) -> conflict-mitigating

// 256 threads = 8 waves (wave32). Wave grid 2(M) x 4(N); each wave owns a
// 64x32 output region = 4 (M) x 2 (N) WMMA 16x16 tiles.
__global__ __launch_bounds__(256)
void w8a16_wmma_kernel(const float* __restrict__ X,
                       const float* __restrict__ Wq,
                       const float* __restrict__ scales,
                       const float* __restrict__ zp,
                       const float* __restrict__ bias,
                       float* __restrict__ out,
                       int M, int N, int K)
{
    __shared__ _Float16 ldsA[BM * SP];
    __shared__ _Float16 ldsB[BN * SP];

    const int tid   = threadIdx.x;
    const int lane  = tid & 31;
    const int wave  = tid >> 5;
    const int waveM = wave >> 2;   // 0..1
    const int waveN = wave & 3;    // 0..3
    const int g     = lane >> 4;   // lane group (0: lanes 0-15, 1: lanes 16-31)
    const int ln    = lane & 15;

    const int blockRow = blockIdx.y * BM;   // M tile base
    const int blockCol = blockIdx.x * BN;   // N tile base

    // Global staging: thread t loads float4 f = t + 256*j of the 128x32 fp32
    // tile; row = f/8, col4 = (f%8)*4. Each thread: 4 float4 per matrix.
    const int r0 = tid >> 3;         // 0..31
    const int c4 = (tid & 7) * 4;    // 0,4,...,28

    const float4* xg[4];
    const float4* wg[4];
    float zpr[4];
#pragma unroll
    for (int j = 0; j < 4; ++j) {
        const int row = r0 + 32 * j;
        xg[j]  = (const float4*)(X  + (size_t)(blockRow + row) * K + c4);
        wg[j]  = (const float4*)(Wq + (size_t)(blockCol + row) * K + c4);
        zpr[j] = zp[blockCol + row];          // per-output-channel zero point
    }

    v8f acc[4][2];
#pragma unroll
    for (int mi = 0; mi < 4; ++mi)
#pragma unroll
        for (int ni = 0; ni < 2; ++ni)
            acc[mi][ni] = (v8f)(0.0f);

    const int kSteps = K / BK;

    float4 xa[4], wa[4];
#pragma unroll
    for (int j = 0; j < 4; ++j) { xa[j] = xg[j][0]; wa[j] = wg[j][0]; }

    for (int kt = 0; kt < kSteps; ++kt) {
        __syncthreads();   // previous tile fully consumed
        // Stage fp32 -> f16 into LDS (weights: exact int8-in-f16; fold zp here)
#pragma unroll
        for (int j = 0; j < 4; ++j) {
            const int row = r0 + 32 * j;
            v4h ha, hb;
            ha[0] = (_Float16)xa[j].x; ha[1] = (_Float16)xa[j].y;
            ha[2] = (_Float16)xa[j].z; ha[3] = (_Float16)xa[j].w;
            hb[0] = (_Float16)(wa[j].x - zpr[j]); hb[1] = (_Float16)(wa[j].y - zpr[j]);
            hb[2] = (_Float16)(wa[j].z - zpr[j]); hb[3] = (_Float16)(wa[j].w - zpr[j]);
            *(v4h*)&ldsA[row * SP + c4] = ha;
            *(v4h*)&ldsB[row * SP + c4] = hb;
        }
        __syncthreads();

        // Prefetch next k-tile while WMMAs run on this one
        if (kt + 1 < kSteps) {
            const int o = (kt + 1) * (BK / 4);
#pragma unroll
            for (int j = 0; j < 4; ++j) { xa[j] = xg[j][o]; wa[j] = wg[j][o]; }
        }

        // B fragments (32x16 f16): lane=n, halves 0-7: K=g*8+h, 8-15: K=16+g*8+(h-8)
        v16h bfrag[2];
#pragma unroll
        for (int ni = 0; ni < 2; ++ni) {
            const int row = waveN * 32 + ni * 16 + ln;   // n within block tile
            const v8h* p = (const v8h*)&ldsB[row * SP + g * 8];
            const v8h lo = p[0], hi = p[2];
#pragma unroll
            for (int e = 0; e < 8; ++e) { bfrag[ni][e] = lo[e]; bfrag[ni][8 + e] = hi[e]; }
        }

        // A fragments (16x32 f16): lane=m, halves 0-7: K=g*8+h, 8-15: K=16+g*8+(h-8)
#pragma unroll
        for (int mi = 0; mi < 4; ++mi) {
            const int row = waveM * 64 + mi * 16 + ln;   // m within block tile
            const v8h lo = *(const v8h*)&ldsA[row * SP + g * 8];
            const v8h hi = *(const v8h*)&ldsA[row * SP + 16 + g * 8];
            v16h afrag;
#pragma unroll
            for (int e = 0; e < 8; ++e) { afrag[e] = lo[e]; afrag[8 + e] = hi[e]; }
#pragma unroll
            for (int ni = 0; ni < 2; ++ni) {
                acc[mi][ni] = __builtin_amdgcn_wmma_f32_16x16x32_f16(
                    /*neg_a=*/false, afrag, /*neg_b=*/false, bfrag[ni],
                    /*c_mod=*/(short)0, acc[mi][ni],
                    /*reuse_a=*/false, /*reuse_b=*/false);
            }
        }
    }

    // Epilogue: C/D layout -> VGPR i holds row (g*8 + i), col = lane%16.
    // Apply per-channel scale post-accumulation, add bias, store fp32.
#pragma unroll
    for (int ni = 0; ni < 2; ++ni) {
        const int colG = blockCol + waveN * 32 + ni * 16 + ln;
        const float s  = scales[colG];
        const float bv = bias[colG];
#pragma unroll
        for (int mi = 0; mi < 4; ++mi) {
            const int rowBase = blockRow + waveM * 64 + mi * 16 + g * 8;
#pragma unroll
            for (int i = 0; i < 8; ++i) {
                out[(size_t)(rowBase + i) * N + colG] = acc[mi][ni][i] * s + bv;
            }
        }
    }
}

extern "C" void kernel_launch(void* const* d_in, const int* in_sizes, int n_in,
                              void* d_out, int out_size, void* d_ws, size_t ws_size,
                              hipStream_t stream) {
    const float* X  = (const float*)d_in[0];   // [8,2048,4096] fp32
    const float* Wq = (const float*)d_in[1];   // [4096,4096] fp32 holding int8 values
    const float* sc = (const float*)d_in[2];   // [4096,1]
    const float* zp = (const float*)d_in[3];   // [4096,1]
    const float* bs = (const float*)d_in[4];   // [1,4096]

    float* out = (float*)d_out;

    const int N = in_sizes[2];           // out_features (scales count)
    const int K = in_sizes[1] / N;       // in_features
    const int M = in_sizes[0] / K;       // 8*2048 flattened rows

    dim3 grid(N / BN, M / BM);           // 32 x 128 blocks
    w8a16_wmma_kernel<<<grid, dim3(256), 0, stream>>>(X, Wq, sc, zp, bs, out, M, N, K);
}
